// GE2ELoss_80126909874352
// MI455X (gfx1250) — matrix operand order repaired and checked by reading, not compiled, not run on hardware
//
#include <hip/hip_runtime.h>
#include <hip/hip_bf16.h>
#include <math.h>

// Problem constants (from reference): N speakers, M utts, D dim
#define NSPK 1024
#define MUTT 32
#define DDIM 512
#define NROWS (NSPK * MUTT)   // 32768

typedef __attribute__((ext_vector_type(16))) _Float16 v16h;
typedef __attribute__((ext_vector_type(8)))  _Float16 v8h;
typedef __attribute__((ext_vector_type(4)))  _Float16 v4h;
typedef __attribute__((ext_vector_type(8)))  float    v8f;
typedef __attribute__((ext_vector_type(4)))  int      v4i;

// ---------------------------------------------------------------------------
// Async global->LDS copy helpers (CDNA5 ASYNCcnt path)
// ---------------------------------------------------------------------------
__device__ __forceinline__ void async_copy_b128(const void* g, void* l) {
#if __has_builtin(__builtin_amdgcn_global_load_async_to_lds_b128)
    __builtin_amdgcn_global_load_async_to_lds_b128(
        (__attribute__((address_space(1))) v4i*)g,
        (__attribute__((address_space(3))) v4i*)l, 0, 0);
#else
    unsigned lofs = (unsigned)(uintptr_t)l;   // LDS aperture: low 32 bits = LDS offset
    asm volatile("global_load_async_to_lds_b128 %0, %1, off"
                 :: "v"(lofs), "v"(g) : "memory");
#endif
}

__device__ __forceinline__ void wait_async0() {
#if __has_builtin(__builtin_amdgcn_s_wait_asynccnt)
    __builtin_amdgcn_s_wait_asynccnt(0);
#else
    asm volatile("s_wait_asynccnt 0x0" ::: "memory");
#endif
}

// ---------------------------------------------------------------------------
// Kernel 1: inclusive centroids ck = mean over M, stored f32 (for exact ex)
// and f16 (WMMA B operand).
// ---------------------------------------------------------------------------
__global__ __launch_bounds__(256) void ge2e_centroid(const float* __restrict__ emb,
                                                     float* __restrict__ ck32,
                                                     _Float16* __restrict__ ck16) {
    int idx = blockIdx.x * blockDim.x + threadIdx.x;   // over NSPK*DDIM
    if (idx >= NSPK * DDIM) return;
    int i = idx / DDIM;
    int d = idx - i * DDIM;
    const float* base = emb + (size_t)i * MUTT * DDIM + d;
    float s = 0.0f;
#pragma unroll
    for (int j = 0; j < MUTT; ++j) s += base[j * DDIM];
    float c = s * (1.0f / MUTT);
    ck32[idx] = c;
    ck16[idx] = (_Float16)c;
}

// ---------------------------------------------------------------------------
// Kernel 2: convert embeddings f32 -> f16 (WMMA A operand), vectorized x4.
// ---------------------------------------------------------------------------
__global__ __launch_bounds__(256) void ge2e_cvt16(const float* __restrict__ emb,
                                                  _Float16* __restrict__ e16) {
    int idx = blockIdx.x * blockDim.x + threadIdx.x;   // over NROWS*DDIM/4
    if (idx >= NROWS * DDIM / 4) return;
    float4 f = ((const float4*)emb)[idx];
    v4h h = { (_Float16)f.x, (_Float16)f.y, (_Float16)f.z, (_Float16)f.w };
    ((v4h*)e16)[idx] = h;
}

// ---------------------------------------------------------------------------
// Kernel 3: exact exclusive-centroid similarity per row (f32), wave per row.
// ex = (M*<e,ck> - <e,e>) / (M-1)
// ---------------------------------------------------------------------------
__global__ __launch_bounds__(256) void ge2e_ex(const float* __restrict__ emb,
                                               const float* __restrict__ ck32,
                                               float* __restrict__ ex) {
    int wave = (blockIdx.x * blockDim.x + threadIdx.x) >> 5;
    int lane = threadIdx.x & 31;
    if (wave >= NROWS) return;
    int sp = wave / MUTT;
    const float* e = emb  + (size_t)wave * DDIM;
    const float* c = ck32 + (size_t)sp   * DDIM;
    float dec = 0.0f, dee = 0.0f;
#pragma unroll 4
    for (int k = lane; k < DDIM; k += 32) {
        float ev = e[k];
        dec += ev * c[k];
        dee += ev * ev;
    }
#pragma unroll
    for (int off = 16; off > 0; off >>= 1) {
        dec += __shfl_xor(dec, off, 32);
        dee += __shfl_xor(dee, off, 32);
    }
    if (lane == 0) ex[wave] = ((float)MUTT * dec - dee) * (1.0f / (MUTT - 1));
}

// ---------------------------------------------------------------------------
// Kernel 4: zero the two global accumulators.
// ---------------------------------------------------------------------------
__global__ void ge2e_init(float* accum) {
    if (threadIdx.x < 2) accum[threadIdx.x] = 0.0f;
}

// ---------------------------------------------------------------------------
// Kernel 5: fused WMMA GEMM + masked row max/argmax + sigmoid + reductions.
// 128 threads = 4 waves; each wave owns 16 rows x all 1024 columns.
// A tile (16x512 f16) register-resident in exact WMMA layout.
// B tiles (16 cols x 512 K = 16 KB) staged in LDS via async global->LDS
// copies, double-buffered so tile ct+1 streams in during the WMMAs of ct.
// B register fetch is software-pipelined (prefetch kk+1 before WMMA kk).
// ---------------------------------------------------------------------------
__global__ __launch_bounds__(128) void ge2e_main(const _Float16* __restrict__ E,
                                                 const _Float16* __restrict__ Ck,
                                                 const float* __restrict__ ex,
                                                 const float* __restrict__ ln_w,
                                                 const float* __restrict__ ln_b,
                                                 float* __restrict__ accum) {
    __shared__ __align__(16) _Float16 lds_b[2][16 * DDIM];   // 2 x 16 KB

    const int tid  = threadIdx.x;            // 0..127
    const int lane = tid & 31;
    const int wave = tid >> 5;
    const int half = lane >> 4;              // K-half selector per WMMA layout
    const int l15  = lane & 15;
    const int r0   = blockIdx.x * 64 + wave * 16;

    // ---- issue async fill of B tile 0 (16 contiguous KB of Ck) ----
    {
        const char* gsrc = (const char*)Ck;              // tile 0 @ col 0
        char* ldst = (char*)&lds_b[0][0];
#pragma unroll
        for (int r = 0; r < 8; ++r) {
            int off = tid * 16 + r * 2048;               // 128 thr x 16 B x 8
            async_copy_b128(gsrc + off, ldst + off);
        }
    }

    // ---- load A (16 rows x 512 K) into registers in WMMA 16x32 layout ----
    // lane<16: elems 0..7 -> K=k0+0..7,  elems 8..15 -> K=k0+16..23
    // lane>=16: elems 0..7 -> K=k0+8..15, elems 8..15 -> K=k0+24..31
    v16h a[16];
    const _Float16* arow = E + (size_t)(r0 + l15) * DDIM;
#pragma unroll
    for (int kk = 0; kk < 16; ++kk) {
        v8h lo = *(const v8h*)(arow + kk * 32 + half * 8);
        v8h hi = *(const v8h*)(arow + kk * 32 + 16 + half * 8);
        a[kk] = __builtin_shufflevector(lo, hi, 0,1,2,3,4,5,6,7,8,9,10,11,12,13,14,15);
    }

    float maxv[8];
    int   maxc[8];
#pragma unroll
    for (int v = 0; v < 8; ++v) { maxv[v] = -3.4e38f; maxc[v] = 0x7fffffff; }

    wait_async0();
    __syncthreads();

    // ---- sweep all 64 column tiles of 16 speakers each ----
    for (int ct = 0; ct < 64; ++ct) {
        const int cur = ct & 1;

        // prefetch next B tile into the other LDS buffer (overlaps WMMAs)
        if (ct + 1 < 64) {
            const char* gsrc = (const char*)(Ck + (size_t)(ct + 1) * 16 * DDIM);
            char* ldst = (char*)&lds_b[cur ^ 1][0];
#pragma unroll
            for (int r = 0; r < 8; ++r) {
                int off = tid * 16 + r * 2048;
                async_copy_b128(gsrc + off, ldst + off);
            }
        }

        // B 32x16 layout: lane holds column c0+l15; lane<16 K=k0+0..7,16..23,
        // lane>=16 K=k0+8..15,24..31 (same K order as A).
        // Software pipeline: fetch b[kk+1] from LDS before WMMA on b[kk].
        const _Float16* brow = &lds_b[cur][l15 * DDIM + half * 8];
        v8f acc = {};
        v8h lo0 = *(const v8h*)(brow);
        v8h hi0 = *(const v8h*)(brow + 16);
#pragma unroll
        for (int kk = 0; kk < 16; ++kk) {
            v8h lo1 = {}, hi1 = {};
            if (kk + 1 < 16) {
                lo1 = *(const v8h*)(brow + (kk + 1) * 32);
                hi1 = *(const v8h*)(brow + (kk + 1) * 32 + 16);
            }
            v16h b = __builtin_shufflevector(lo0, hi0, 0,1,2,3,4,5,6,7,8,9,10,11,12,13,14,15);
            acc = __builtin_amdgcn_wmma_f32_16x16x32_f16(
                false, a[kk], false, b, (short)0, acc, false, false);
            lo0 = lo1; hi0 = hi1;
        }

        // fold this 16x16 block into running masked max/argmax
        const int C = ct * 16 + l15;
#pragma unroll
        for (int v = 0; v < 8; ++v) {
            int R  = r0 + v + half * 8;     // row this element belongs to
            int sp = R >> 5;                // speaker = row / 32
            float val = acc[v];
            bool better = (C != sp) &&
                          ((val > maxv[v]) || (val == maxv[v] && C < maxc[v]));
            if (better) { maxv[v] = val; maxc[v] = C; }
        }

        if (ct + 1 < 64) {
            wait_async0();      // next tile landed in LDS
            __syncthreads();    // all waves done reading cur, next buffer ready
        }
    }

    // ---- cross-lane argmax reduction within each 16-lane half ----
#pragma unroll
    for (int v = 0; v < 8; ++v) {
#pragma unroll
        for (int off = 1; off < 16; off <<= 1) {
            float ov = __shfl_xor(maxv[v], off, 32);
            int   oc = __shfl_xor(maxc[v], off, 32);
            if ((ov > maxv[v]) || (ov == maxv[v] && oc < maxc[v])) {
                maxv[v] = ov; maxc[v] = oc;
            }
        }
    }

    // ---- finalize 8 rows per writing lane (lane 0 and lane 16) ----
    if (l15 == 0) {
        float w  = *ln_w;
        float bb = *ln_b;
        float lsum = 0.0f, csum = 0.0f;
#pragma unroll
        for (int v = 0; v < 8; ++v) {
            int R  = r0 + v + half * 8;
            int sp = R >> 5;
            float exv   = ex[R];
            float own_s = 1.0f / (1.0f + __expf(-(w * exv     + bb)));
            float mo_s  = 1.0f / (1.0f + __expf(-(w * maxv[v] + bb)));
            lsum += 1.0f - own_s + mo_s;
            // sigmoid monotone (ln_w > 0): argmax over raw sims w/ own col = ex
            bool ownwins = (exv > maxv[v]) || (exv == maxv[v] && sp < maxc[v]);
            csum += ownwins ? 1.0f : 0.0f;
        }
        atomicAdd(accum + 0, lsum);
        atomicAdd(accum + 1, csum);
    }
}

// ---------------------------------------------------------------------------
// Kernel 6: finalize means into d_out.
// ---------------------------------------------------------------------------
__global__ void ge2e_finalize(const float* __restrict__ accum, float* __restrict__ out) {
    if (threadIdx.x == 0) {
        out[0] = accum[0] * (1.0f / NROWS);   // loss mean
        out[1] = accum[1] * (1.0f / NROWS);   // batch accuracy
    }
}

// ---------------------------------------------------------------------------
extern "C" void kernel_launch(void* const* d_in, const int* in_sizes, int n_in,
                              void* d_out, int out_size, void* d_ws, size_t ws_size,
                              hipStream_t stream) {
    const float* emb  = (const float*)d_in[0];   // [N, M, D] f32
    const float* ln_w = (const float*)d_in[3];   // scalar
    const float* ln_b = (const float*)d_in[4];   // scalar
    float* out = (float*)d_out;

    // workspace layout (bytes)
    char* ws = (char*)d_ws;
    _Float16* e16  = (_Float16*)(ws + 0);                       // 33,554,432 B
    _Float16* ck16 = (_Float16*)(ws + 33554432);                //  1,048,576 B
    float*    ck32 = (float*)   (ws + 34603008);                //  2,097,152 B
    float*    exbf = (float*)   (ws + 36700160);                //    131,072 B
    float*    acc  = (float*)   (ws + 36831232);                //          8 B

    // 1) centroids (f32 + f16)
    ge2e_centroid<<<(NSPK * DDIM + 255) / 256, 256, 0, stream>>>(emb, ck32, ck16);
    // 2) E -> f16
    ge2e_cvt16<<<(NROWS * DDIM / 4 + 255) / 256, 256, 0, stream>>>(emb, e16);
    // 3) exact exclusive similarity per row (wave per row, 8 waves/block)
    ge2e_ex<<<NROWS / 8, 256, 0, stream>>>(emb, ck32, exbf);
    // 4) zero accumulators
    ge2e_init<<<1, 32, 0, stream>>>(acc);
    // 5) fused WMMA GEMM + reductions: 512 blocks x 4 waves x 16 rows
    ge2e_main<<<NROWS / 64, 128, 0, stream>>>(e16, ck16, exbf, ln_w, ln_b, acc);
    // 6) means
    ge2e_finalize<<<1, 32, 0, stream>>>(acc, out);
}
